// AnomalyTransformerModel_86294482911877
// MI455X (gfx1250) — matrix operand order repaired and checked by reading, not compiled, not run on hardware
//
#include <hip/hip_runtime.h>
#include <math.h>

// Problem constants (match reference)
#define Bz    8
#define Lz    512
#define CINz  55
#define Dz    512
#define Hz    8
#define NLz   3
#define DFFz  2048
#define COUTz 55
#define Ez    64
#define Mz    (Bz * Lz)   // 4096 rows for all token GEMMs

typedef __attribute__((ext_vector_type(16))) __bf16 v16bf;
typedef __attribute__((ext_vector_type(8)))  __bf16 v8bf;
typedef __attribute__((ext_vector_type(8)))  float  v8f;

static __device__ __forceinline__ __bf16 to_bf16(float f) {
    return (__bf16)f;   // native v_cvt on gfx1250
}

// combine two contiguous 8-element bf16 chunks into a WMMA fragment
static __device__ __forceinline__ v16bf frag16(const __bf16* p0, const __bf16* p1) {
    v8bf a = *(const v8bf*)p0;
    v8bf b = *(const v8bf*)p1;
    return __builtin_shufflevector(a, b, 0, 1, 2, 3, 4, 5, 6, 7,
                                   8, 9, 10, 11, 12, 13, 14, 15);
}

// async copy of one 16-byte chunk global -> LDS (CDNA5, tracked by ASYNCcnt)
static __device__ __forceinline__ void async_copy_b128(const __bf16* gsrc, __bf16* ldst) {
    unsigned           dst = (unsigned)(unsigned long long)(void*)ldst; // LDS offset = low 32 bits
    unsigned long long src = (unsigned long long)gsrc;
    asm volatile("global_load_async_to_lds_b128 %0, %1, off"
                 :: "v"(dst), "v"(src) : "memory");
}

static __device__ __forceinline__ void wait_async_all() {
    asm volatile("s_wait_asynccnt 0x0" ::: "memory");
}

static __device__ __forceinline__ float gelu_exact(float x) {
    return 0.5f * x * (1.0f + erff(x * 0.7071067811865475f));
}

// ---------------------------------------------------------------------------
// One-time: W[K,N] f32  ->  Wt[Npad,K] bf16 (transposed, zero-padded rows)
// grid = (K/32, Npad/32), block = (32,8)
// ---------------------------------------------------------------------------
__global__ void transpose_conv_bf16(const float* __restrict__ W,
                                    __bf16* __restrict__ Wt,
                                    int K, int N) {
    __shared__ float tile[32][33];
    const int k0 = blockIdx.x * 32;
    const int n0 = blockIdx.y * 32;
    const int tx = threadIdx.x;   // 0..31
    const int ty = threadIdx.y;   // 0..7
    #pragma unroll
    for (int r = 0; r < 4; ++r) {
        int k = k0 + ty + r * 8;
        int n = n0 + tx;
        tile[ty + r * 8][tx] = (n < N) ? W[(size_t)k * N + n] : 0.f;
    }
    __syncthreads();
    #pragma unroll
    for (int r = 0; r < 4; ++r) {
        int n = n0 + ty + r * 8;          // row of Wt
        int k = k0 + tx;
        Wt[(size_t)n * K + k] = to_bf16(tile[tx][ty + r * 8]);
    }
}

// ---------------------------------------------------------------------------
// Embedding: circular conv1d(k=3) + sinusoidal PE -> h (f32) and h_bf (bf16)
// ---------------------------------------------------------------------------
__global__ void embed_kernel(const float* __restrict__ x,
                             const float* __restrict__ conv_w,
                             float* __restrict__ h,
                             __bf16* __restrict__ h_bf) {
    const int bt = blockIdx.x;          // b*L + t
    const int b  = bt / Lz;
    const int t  = bt % Lz;
    __shared__ float xw[3 * CINz];
    if (threadIdx.x < 3 * CINz) {
        int kk = threadIdx.x / CINz;
        int c  = threadIdx.x % CINz;
        int tt = (t + kk - 1 + Lz) % Lz;   // wrap padding
        xw[threadIdx.x] = x[(b * Lz + tt) * CINz + c];
    }
    __syncthreads();
    for (int d = threadIdx.x; d < Dz; d += blockDim.x) {
        float s = 0.f;
        #pragma unroll
        for (int kk = 0; kk < 3; ++kk) {
            const float* wrow = conv_w + (d * CINz) * 3 + kk;
            const float* xrow = xw + kk * CINz;
            for (int c = 0; c < CINz; ++c)
                s += xrow[c] * wrow[c * 3];
        }
        int   i2  = d >> 1;
        float div = expf((float)(2 * i2) * (-9.210340371976184f / (float)Dz));
        float ang = (float)t * div;
        s += (d & 1) ? cosf(ang) : sinf(ang);
        h[(size_t)bt * Dz + d]    = s;
        h_bf[(size_t)bt * Dz + d] = to_bf16(s);
    }
}

// ---------------------------------------------------------------------------
// bf16 WMMA GEMM: val = act(A[M,K] @ Wt[N,K]^T + bias[N]) (+ resid[M,N])
//   A  : bf16 [M,K]      Wt : bf16 [Npad,K] (pre-transposed, padded)
//   out (f32, nullable) / out_bf (bf16, nullable) / out_bf_t (bf16 [B,D,L])
// block = 128 threads (4 waves), tile 64x64, K stepped by 32.
// Tile staging via async global->LDS b128 copies (ASYNCcnt).
// ---------------------------------------------------------------------------
__global__ __launch_bounds__(128)
void gemm_bf16_wmma(const __bf16* __restrict__ A,
                    const __bf16* __restrict__ Wt,
                    const float* __restrict__ bias,
                    const float* __restrict__ resid,
                    float* __restrict__ out,
                    __bf16* __restrict__ out_bf,
                    __bf16* __restrict__ out_bf_t,
                    int M, int N, int K, int act) {
    __shared__ __attribute__((aligned(16))) __bf16 sA [64 * 32];  // [row][k]
    __shared__ __attribute__((aligned(16))) __bf16 sWt[64 * 32];  // [n][k]

    const int tid  = threadIdx.x;
    const int wave = tid >> 5;
    const int lane = tid & 31;
    const int m0   = blockIdx.y * 64;
    const int n0   = blockIdx.x * 64;

    v8f acc[4];
    #pragma unroll
    for (int j = 0; j < 4; ++j)
        acc[j] = (v8f){0.f, 0.f, 0.f, 0.f, 0.f, 0.f, 0.f, 0.f};

    const int hi  = lane >> 4;        // 0 or 1
    const int lo  = lane & 15;
    const int kbA = hi * 8;           // A fragment K base
    const int kbB = hi * 16;          // unused; B fragment also uses kbA

    // per-thread staging chunk coordinates (2 chunks of 16B each per tile)
    for (int k0 = 0; k0 < K; k0 += 32) {
        #pragma unroll
        for (int i = 0; i < 2; ++i) {
            int c   = i * 128 + tid;        // 256 chunks per tile
            int row = c >> 2;
            int cc  = (c & 3) * 8;
            async_copy_b128(&A [(size_t)(m0 + row) * K + k0 + cc], &sA [row * 32 + cc]);
            async_copy_b128(&Wt[(size_t)(n0 + row) * K + k0 + cc], &sWt[row * 32 + cc]);
        }
        if (k0 + 32 < K) {  // prefetch next k-step into caches
            __builtin_prefetch(&A [(size_t)(m0 + (tid >> 1)) * K + k0 + 32], 0, 3);
            __builtin_prefetch(&Wt[(size_t)(n0 + (tid >> 1)) * K + k0 + 32], 0, 3);
        }
        wait_async_all();
        __syncthreads();

        const int mrow = wave * 16 + lo;
        v16bf af = frag16(&sA[mrow * 32 + kbA], &sA[mrow * 32 + kbA + 16]);
        #pragma unroll
        for (int j = 0; j < 4; ++j) {
            int nn = j * 16 + lo;
            v16bf bf = frag16(&sWt[nn * 32 + kbA], &sWt[nn * 32 + kbA + 16]);
            acc[j] = __builtin_amdgcn_wmma_f32_16x16x32_bf16(
                false, af, false, bf, (short)0, acc[j], false, false);
        }
        __syncthreads();
    }

    // store: C layout -> VGPR r: M = r + 8*hi, N = lo
    #pragma unroll
    for (int j = 0; j < 4; ++j) {
        int gcol = n0 + j * 16 + lo;
        if (gcol < N) {
            #pragma unroll
            for (int r = 0; r < 8; ++r) {
                int grow  = m0 + wave * 16 + r + hi * 8;
                float val = acc[j][r] + bias[gcol];
                if (act) val = gelu_exact(val);
                if (resid) val += resid[(size_t)grow * N + gcol];
                if (out)    out[(size_t)grow * N + gcol] = val;
                if (out_bf) out_bf[(size_t)grow * N + gcol] = to_bf16(val);
                if (out_bf_t) {
                    // [B, D, L]: ((b*D + gcol) * L + srow); N must be 512
                    int b    = grow >> 9;
                    int srow = grow & (Lz - 1);
                    out_bf_t[((size_t)(b * Dz + gcol)) * Lz + srow] = to_bf16(val);
                }
            }
        }
    }
}

// ---------------------------------------------------------------------------
// Sigma projection: sig_raw[b,t,h] = q[b,t,:] . Ws[:,h] + bs[h]   (N=8, plain)
// ---------------------------------------------------------------------------
__global__ void sigma_kernel(const float* __restrict__ q,
                             const float* __restrict__ Ws,
                             const float* __restrict__ bs,
                             float* __restrict__ sig_raw) {
    int idx = blockIdx.x * blockDim.x + threadIdx.x;   // [0, B*L*H)
    if (idx >= Mz * Hz) return;
    int hh = idx % Hz;
    int m  = idx / Hz;
    float s = bs[hh];
    const float* qr = q + (size_t)m * Dz;
    for (int d = 0; d < Dz; ++d)
        s += qr[d] * Ws[d * Hz + hh];
    sig_raw[idx] = s;
}

// ---------------------------------------------------------------------------
// Fused attention for one layer (bf16 operands):
//   scores = scale * Qh Kh^T (WMMA) -> softmax -> series/prior/sigma out
//   attn   = series @ Vh     (WMMA, V in [B,D,L] transposed layout) -> bf16
// grid = B*H*(L/16) blocks of 128 threads; block owns one 16-row stripe.
// ---------------------------------------------------------------------------
__global__ __launch_bounds__(128)
void attn_kernel(const __bf16* __restrict__ q_bf,   // [M, D]
                 const __bf16* __restrict__ k_bf,   // [M, D]
                 const __bf16* __restrict__ v_bf_t, // [B, D, L]
                 const float*  __restrict__ sig_raw,
                 float* __restrict__ series_out,
                 float* __restrict__ prior_out,
                 float* __restrict__ sigma_out,
                 __bf16* __restrict__ attn_bf) {    // [M, D]
    __shared__ __attribute__((aligned(16))) float  s_scores[16 * Lz]; // 32 KB
    __shared__ __attribute__((aligned(16))) __bf16 s_series[16 * Lz]; // 16 KB
    __shared__ float s_red[16][8];
    __shared__ float s_rowmax[16];
    __shared__ float s_rowsum[16];
    __shared__ float s_sigma[16];

    const int tid  = threadIdx.x;
    const int wave = tid >> 5;
    const int lane = tid & 31;
    const int hi   = lane >> 4;
    const int lo   = lane & 15;
    const int kbA  = hi * 8;
    const int kbB  = hi * 16;

    const int rb = blockIdx.x & 31;        // row block (L/16 = 32)
    const int bh = blockIdx.x >> 5;
    const int b  = bh >> 3;
    const int hh = bh & 7;
    const float scale = 0.125f;            // 1/sqrt(E=64)

    // ---- Phase 1: scores[16 x 512] = Q(16x64) @ K^T via WMMA -------------
    v16bf af[2];
    {
        const int mrow = rb * 16 + lo;
        const __bf16* qp = q_bf + ((size_t)(b * Lz + mrow)) * Dz + hh * Ez;
        #pragma unroll
        for (int s = 0; s < 2; ++s)
            af[s] = frag16(qp + s * 32 + kbA, qp + s * 32 + kbA + 16);
    }
    #pragma unroll
    for (int j = 0; j < 8; ++j) {
        v8f acc = (v8f){0.f, 0.f, 0.f, 0.f, 0.f, 0.f, 0.f, 0.f};
        int ncol = wave * 128 + j * 16 + lo;
        const __bf16* kp = k_bf + ((size_t)(b * Lz + ncol)) * Dz + hh * Ez;
        #pragma unroll
        for (int s = 0; s < 2; ++s) {
            v16bf bf = frag16(kp + s * 32 + kbA, kp + s * 32 + kbA + 16);
            acc = __builtin_amdgcn_wmma_f32_16x16x32_bf16(
                false, af[s], false, bf, (short)0, acc, false, false);
        }
        #pragma unroll
        for (int r = 0; r < 8; ++r) {
            int row = r + hi * 8;
            int col = wave * 128 + j * 16 + lo;
            s_scores[row * Lz + col] = acc[r] * scale;
        }
    }
    __syncthreads();

    // ---- Phase 2: parallel row softmax (8 threads per row) ----------------
    {
        int row = tid >> 3;
        int seg = tid & 7;
        const float* rowp = s_scores + row * Lz;
        float mx = -3.402823e38f;
        for (int c = seg * 64; c < seg * 64 + 64; ++c) mx = fmaxf(mx, rowp[c]);
        s_red[row][seg] = mx;
        __syncthreads();
        if (seg == 0) {
            float m2 = s_red[row][0];
            #pragma unroll
            for (int i = 1; i < 8; ++i) m2 = fmaxf(m2, s_red[row][i]);
            s_rowmax[row] = m2;
        }
        __syncthreads();
        float rmax = s_rowmax[row];
        float* rowm = s_scores + row * Lz;
        float sum = 0.f;
        for (int c = seg * 64; c < seg * 64 + 64; ++c) {
            float e = expf(rowm[c] - rmax);
            rowm[c] = e;
            sum += e;
        }
        s_red[row][seg] = sum;
        __syncthreads();
        if (seg == 0) {
            float s2 = 0.f;
            #pragma unroll
            for (int i = 0; i < 8; ++i) s2 += s_red[row][i];
            s_rowsum[row] = s2;
        }
        if (tid < 16) {
            // sigma: sigmoid(5x)+1e-5 -> 3^x - 1
            float sv = sig_raw[((size_t)(b * Lz + rb * 16 + tid)) * Hz + hh];
            float xx = 1.0f / (1.0f + expf(-5.0f * sv)) + 1e-5f;
            s_sigma[tid] = expf(xx * 1.0986122886681098f) - 1.0f;   // ln(3)
        }
    }
    __syncthreads();

    // ---- Phase 3a: write series / prior / sigma, stash series as bf16 ----
    for (int idx = tid; idx < 16 * Lz; idx += 128) {
        int row = idx >> 9;
        int col = idx & (Lz - 1);
        int row_g = rb * 16 + row;
        float p = s_scores[idx] / s_rowsum[row];
        s_series[idx] = to_bf16(p);
        float sg   = s_sigma[row];
        float dist = fabsf((float)row_g - (float)col);
        float pr   = expf(-dist * dist / (2.0f * sg * sg)) * (0.3989422804014327f / sg);
        size_t off = (((size_t)(b * Hz + hh)) * Lz + row_g) * Lz + col;
        series_out[off] = p;
        prior_out[off]  = pr;
        sigma_out[off]  = sg;
    }
    __syncthreads();

    // ---- Phase 3b: attn[16 x 64] = series[16 x 512] @ V[512 x 64] --------
    {
        v8f acc = (v8f){0.f, 0.f, 0.f, 0.f, 0.f, 0.f, 0.f, 0.f};
        const int ecol = wave * 16 + lo;
        const __bf16* vt = v_bf_t + ((size_t)(b * Dz + hh * Ez + ecol)) * Lz;
        #pragma unroll
        for (int k0 = 0; k0 < Lz; k0 += 32) {
            const __bf16* sp = s_series + lo * Lz + k0 + kbA;
            v16bf pa = frag16(sp, sp + 16);
            v16bf vb = frag16(vt + k0 + kbA, vt + k0 + kbA + 16);
            acc = __builtin_amdgcn_wmma_f32_16x16x32_bf16(
                false, pa, false, vb, (short)0, acc, false, false);
        }
        #pragma unroll
        for (int r = 0; r < 8; ++r) {
            int row_g = rb * 16 + r + hi * 8;
            attn_bf[((size_t)(b * Lz + row_g)) * Dz + hh * Ez + ecol] = to_bf16(acc[r]);
        }
    }
}

// ---------------------------------------------------------------------------
// Host-side orchestration
// ---------------------------------------------------------------------------
extern "C" void kernel_launch(void* const* d_in, const int* in_sizes, int n_in,
                              void* d_out, int out_size, void* d_ws, size_t ws_size,
                              hipStream_t stream) {
    (void)in_sizes; (void)n_in; (void)out_size; (void)ws_size;

    const float* x      = (const float*)d_in[0];
    const float* conv_w = (const float*)d_in[1];
    const float* Wq     = (const float*)d_in[2];
    const float* bq     = (const float*)d_in[3];
    const float* Wk     = (const float*)d_in[4];
    const float* bk     = (const float*)d_in[5];
    const float* Wv     = (const float*)d_in[6];
    const float* bv     = (const float*)d_in[7];
    const float* Ws     = (const float*)d_in[8];
    const float* bs     = (const float*)d_in[9];
    const float* Wo     = (const float*)d_in[10];
    const float* bo     = (const float*)d_in[11];
    const float* W1     = (const float*)d_in[12];
    const float* b1     = (const float*)d_in[13];
    const float* W2     = (const float*)d_in[14];
    const float* b2     = (const float*)d_in[15];
    const float* Wp     = (const float*)d_in[16];
    const float* bp     = (const float*)d_in[17];

    // output layout: out [B,L,COUT] | series | prior | sigma (each [NL,B,H,L,L])
    float* out_p    = (float*)d_out;
    float* series_p = out_p + (size_t)Bz * Lz * COUTz;
    float* prior_p  = series_p + (size_t)NLz * Bz * Hz * Lz * Lz;
    float* sigma_p  = prior_p  + (size_t)NLz * Bz * Hz * Lz * Lz;

    // workspace carve: f32 region then bf16 region
    const size_t MD = (size_t)Mz * Dz;               // 2M elements
    float* ws   = (float*)d_ws;
    float* h    = ws;                                // [M, D]  f32
    float* qb   = h  + MD;                           // [M, D]  f32 (for sigma)
    float* sig  = qb + MD;                           // [M, H]  f32

    __bf16* bfp    = (__bf16*)(sig + (size_t)Mz * Hz);
    __bf16* h_bf   = bfp;              bfp += MD;            // [M, D]
    __bf16* q_bf   = bfp;              bfp += MD;            // [M, D]
    __bf16* k_bf   = bfp;              bfp += MD;            // [M, D]
    __bf16* v_bf_t = bfp;              bfp += MD;            // [B, D, L]
    __bf16* attn_bf= bfp;              bfp += MD;            // [M, D]
    __bf16* y1_bf  = bfp;              bfp += (size_t)Mz * DFFz; // [M, DFF]
    // pre-transposed bf16 weights
    const size_t WDD = (size_t)Dz * Dz, WDF = (size_t)Dz * DFFz;
    __bf16* Wqt = bfp;                 bfp += WDD * NLz;
    __bf16* Wkt = bfp;                 bfp += WDD * NLz;
    __bf16* Wvt = bfp;                 bfp += WDD * NLz;
    __bf16* Wot = bfp;                 bfp += WDD * NLz;
    __bf16* W1t = bfp;                 bfp += WDF * NLz;     // [2048, 512]
    __bf16* W2t = bfp;                 bfp += WDF * NLz;     // [512, 2048]
    __bf16* Wpt = bfp;                                        // [64, 512]

    const dim3 tblk(32, 8);
    // one-time weight transpose+convert (runs inside the graph; deterministic)
    for (int l = 0; l < NLz; ++l) {
        transpose_conv_bf16<<<dim3(Dz / 32, Dz / 32), tblk, 0, stream>>>(
            Wq + (size_t)l * WDD, Wqt + (size_t)l * WDD, Dz, Dz);
        transpose_conv_bf16<<<dim3(Dz / 32, Dz / 32), tblk, 0, stream>>>(
            Wk + (size_t)l * WDD, Wkt + (size_t)l * WDD, Dz, Dz);
        transpose_conv_bf16<<<dim3(Dz / 32, Dz / 32), tblk, 0, stream>>>(
            Wv + (size_t)l * WDD, Wvt + (size_t)l * WDD, Dz, Dz);
        transpose_conv_bf16<<<dim3(Dz / 32, Dz / 32), tblk, 0, stream>>>(
            Wo + (size_t)l * WDD, Wot + (size_t)l * WDD, Dz, Dz);
        transpose_conv_bf16<<<dim3(Dz / 32, DFFz / 32), tblk, 0, stream>>>(
            W1 + (size_t)l * WDF, W1t + (size_t)l * WDF, Dz, DFFz);
        transpose_conv_bf16<<<dim3(DFFz / 32, Dz / 32), tblk, 0, stream>>>(
            W2 + (size_t)l * WDF, W2t + (size_t)l * WDF, DFFz, Dz);
    }
    transpose_conv_bf16<<<dim3(Dz / 32, 64 / 32), tblk, 0, stream>>>(Wp, Wpt, Dz, COUTz);

    const dim3 blk128(128);
    const dim3 gD (Dz   / 64, Mz / 64);   // N=512
    const dim3 gF1(DFFz / 64, Mz / 64);   // N=2048
    const dim3 gP (1,         Mz / 64);   // N=55 (padded to 64)

    embed_kernel<<<Mz, 256, 0, stream>>>(x, conv_w, h, h_bf);

    for (int l = 0; l < NLz; ++l) {
        const float* bq_l = bq + (size_t)l * Dz;
        const float* bk_l = bk + (size_t)l * Dz;
        const float* bv_l = bv + (size_t)l * Dz;
        const float* bo_l = bo + (size_t)l * Dz;
        const float* b1_l = b1 + (size_t)l * DFFz;
        const float* b2_l = b2 + (size_t)l * Dz;
        const float* Ws_l = Ws + (size_t)l * Dz * Hz;
        const float* bs_l = bs + (size_t)l * Hz;

        // q: f32 (for sigma proj) + bf16 copy
        gemm_bf16_wmma<<<gD, blk128, 0, stream>>>(h_bf, Wqt + (size_t)l * WDD, bq_l,
                                                  nullptr, qb, q_bf, nullptr,
                                                  Mz, Dz, Dz, 0);
        // k: bf16 only
        gemm_bf16_wmma<<<gD, blk128, 0, stream>>>(h_bf, Wkt + (size_t)l * WDD, bk_l,
                                                  nullptr, nullptr, k_bf, nullptr,
                                                  Mz, Dz, Dz, 0);
        // v: transposed bf16 only
        gemm_bf16_wmma<<<gD, blk128, 0, stream>>>(h_bf, Wvt + (size_t)l * WDD, bv_l,
                                                  nullptr, nullptr, nullptr, v_bf_t,
                                                  Mz, Dz, Dz, 0);
        sigma_kernel<<<(Mz * Hz + 255) / 256, 256, 0, stream>>>(qb, Ws_l, bs_l, sig);

        float* ser_l = series_p + (size_t)l * Bz * Hz * Lz * Lz;
        float* pri_l = prior_p  + (size_t)l * Bz * Hz * Lz * Lz;
        float* sig_l = sigma_p  + (size_t)l * Bz * Hz * Lz * Lz;
        attn_kernel<<<Bz * Hz * (Lz / 16), blk128, 0, stream>>>(
            q_bf, k_bf, v_bf_t, sig, ser_l, pri_l, sig_l, attn_bf);

        // h = h + attn @ Wo + bo   (also refresh h_bf)
        gemm_bf16_wmma<<<gD, blk128, 0, stream>>>(attn_bf, Wot + (size_t)l * WDD, bo_l,
                                                  h, h, h_bf, nullptr,
                                                  Mz, Dz, Dz, 0);
        // y1 = gelu(h @ W1 + b1)  (bf16 only)
        gemm_bf16_wmma<<<gF1, blk128, 0, stream>>>(h_bf, W1t + (size_t)l * WDF, b1_l,
                                                   nullptr, nullptr, y1_bf, nullptr,
                                                   Mz, DFFz, Dz, 1);
        // h = h + y1 @ W2 + b2    (also refresh h_bf)
        gemm_bf16_wmma<<<gD, blk128, 0, stream>>>(y1_bf, W2t + (size_t)l * WDF, b2_l,
                                                  h, h, h_bf, nullptr,
                                                  Mz, Dz, DFFz, 0);
    }

    // out = h @ Wp + bp   (N = 55, tail-guarded on store)
    gemm_bf16_wmma<<<gP, blk128, 0, stream>>>(h_bf, Wpt, bp, nullptr,
                                              out_p, nullptr, nullptr,
                                              Mz, COUTz, Dz, 0);
}
